// Head_44547400794545
// MI455X (gfx1250) — matrix operand order repaired and checked by reading, not compiled, not run on hardware
//
#include <hip/hip_runtime.h>
#include <hip/hip_bf16.h>
#include <math.h>

// ---------------------------------------------------------------------------
// Types for CDNA5 WMMA (wave32): 16x16x32 bf16, f32 accumulate.
// ---------------------------------------------------------------------------
typedef __attribute__((ext_vector_type(16))) __bf16        v16bf;
typedef __attribute__((ext_vector_type(2)))  __bf16        v2bf;
typedef __attribute__((ext_vector_type(8)))  float         v8f;
typedef __attribute__((ext_vector_type(8)))  unsigned int  v8u;   // 16 packed bf16

__device__ __forceinline__ v8f wmma_bf16(v8u a, v8u b, v8f c) {
  // (neg_a, A, neg_b, B, c_mod, C, reuse_a, reuse_b)
  return __builtin_amdgcn_wmma_f32_16x16x32_bf16(
      false, __builtin_bit_cast(v16bf, a),
      false, __builtin_bit_cast(v16bf, b),
      (short)0, c, false, false);
}

__device__ __forceinline__ unsigned short f2bf(float f) {
  unsigned int u = __float_as_uint(f);
  u += 0x7FFFu + ((u >> 16) & 1u);     // round-to-nearest-even
  return (unsigned short)(u >> 16);
}

// Packed f32x2 -> bf16x2 (v_cvt_pk_bf16_f32 when the builtin exists).
__device__ __forceinline__ unsigned int f2bf2(float lo, float hi) {
#if __has_builtin(__builtin_amdgcn_cvt_pk_bf16_f32)
  v2bf r = __builtin_amdgcn_cvt_pk_bf16_f32(lo, hi);
  return __builtin_bit_cast(unsigned int, r);
#else
  return (unsigned int)f2bf(lo) | ((unsigned int)f2bf(hi) << 16);
#endif
}

// Async global->LDS 16-byte copy (GLOBAL_LOAD_ASYNC_TO_LDS_B128, ASYNCcnt).
// Probe-discovered signature: 4 args, param 1 = addrspace(1) "int4"
// (gcc vector_size(16)) pointer, i.e. (global_src, lds_dst, imm_off, cpol).
#if __has_builtin(__builtin_amdgcn_global_load_async_to_lds_b128)
#define HAS_ASYNC_LDS 1
typedef int b128_t __attribute__((vector_size(16)));
typedef __attribute__((address_space(1))) b128_t* gb128_t;
typedef __attribute__((address_space(3))) b128_t* lb128_t;
#else
#define HAS_ASYNC_LDS 0
#endif

__device__ __forceinline__ void async_cp_b128(const void* g, void* l) {
#if HAS_ASYNC_LDS
  __builtin_amdgcn_global_load_async_to_lds_b128(
      (gb128_t)(unsigned long long)(uintptr_t)g,
      (lb128_t)(unsigned int)(uintptr_t)l,   // low 32 bits of flat = LDS offset
      0, 0);
#else
  *(uint4*)l = *(const uint4*)g;
#endif
}

__device__ __forceinline__ void wait_async_lds() {
#if HAS_ASYNC_LDS
#if __has_builtin(__builtin_amdgcn_s_wait_asynccnt)
  __builtin_amdgcn_s_wait_asynccnt(0);
#else
  asm volatile("s_wait_asynccnt 0" ::: "memory");
#endif
#endif
}

#define N_EMB 1024
#define HEAD  128
#define SEQ   2048
#define BATCH 8

// ---------------------------------------------------------------------------
// Kernel 1: fused bf16-converting projection GEMM.
// grid.x: 16384/128 row tiles, grid.y: 0=Q 1=K 2=V. 256 threads = 8 waves.
// Each wave owns a 16x128 output strip (8 accumulator tiles).
// ---------------------------------------------------------------------------
__global__ __launch_bounds__(256) void proj_kernel(
    const float* __restrict__ x,
    const float* __restrict__ Wq, const float* __restrict__ Wk,
    const float* __restrict__ Wv,
    unsigned short* __restrict__ qo, unsigned short* __restrict__ ko,
    unsigned short* __restrict__ vo)
{
  __shared__ unsigned short Xs[128 * 32];   // x tile, row-major  [row][k]
  __shared__ unsigned short Ws[128 * 32];   // W tile, TRANSPOSED [n][k]

  const int which = blockIdx.y;
  const float* W = (which == 0) ? Wq : (which == 1) ? Wk : Wv;
  unsigned short* out = (which == 0) ? qo : (which == 1) ? ko : vo;

  const int rbase = blockIdx.x * 128;
  const int tid  = threadIdx.x;
  const int wave = tid >> 5;
  const int lane = tid & 31;
  const int lhalf = lane >> 4;
  const int lcol  = lane & 15;

  v8f acc[8] = {};

  for (int k0 = 0; k0 < N_EMB; k0 += 32) {
    // --- stage x tile (128x32 f32 -> bf16, row-major, packed stores) ---
#pragma unroll
    for (int i = 0; i < 4; ++i) {
      int s   = tid + i * 256;           // 1024 float4 slots
      int row = s >> 3;
      int cc  = (s & 7) * 4;
      float4 f = *(const float4*)(x + (size_t)(rbase + row) * N_EMB + k0 + cc);
      unsigned int* xw = (unsigned int*)Xs;
      xw[(row * 32 + cc) >> 1]     = f2bf2(f.x, f.y);
      xw[(row * 32 + cc + 2) >> 1] = f2bf2(f.z, f.w);
    }
    // --- stage W tile (32x128 f32 -> bf16, transposed to [n][k]) ---
#pragma unroll
    for (int i = 0; i < 4; ++i) {
      int s  = tid + i * 256;            // 1024 float4 slots
      int kr = s >> 5;
      int cc = (s & 31) * 4;
      float4 f = *(const float4*)(W + (size_t)(k0 + kr) * HEAD + cc);
      Ws[(cc + 0) * 32 + kr] = f2bf(f.x);
      Ws[(cc + 1) * 32 + kr] = f2bf(f.y);
      Ws[(cc + 2) * 32 + kr] = f2bf(f.z);
      Ws[(cc + 3) * 32 + kr] = f2bf(f.w);
    }
    __syncthreads();

    // --- A fragment: 16x32 bf16, lanes 0-15 rows, K pairs contiguous ---
    v8u a;
    {
      const unsigned int* xp = (const unsigned int*)Xs;
      int base = (wave * 16 + lcol) * 32;
      int khalf = lhalf * 8;
#pragma unroll
      for (int e2 = 0; e2 < 8; ++e2) {
        int kk = (e2 & 3) * 2 + (e2 >> 2) * 16 + khalf;
        a[e2] = xp[(base + kk) >> 1];
      }
    }
    // --- 8 WMMA tiles across N=128 ---
#pragma unroll
    for (int nt = 0; nt < 8; ++nt) {
      v8u bfrag;
      const unsigned int* wp = (const unsigned int*)Ws;
      int n  = lcol + nt * 16;
      int kh = lhalf * 8;
#pragma unroll
      for (int e2 = 0; e2 < 8; ++e2)
        bfrag[e2] = wp[(n * 32 + kh + (e2 & 3) * 2 + (e2 >> 2) * 16) >> 1];
      acc[nt] = wmma_bf16(a, bfrag, acc[nt]);
    }
    __syncthreads();
  }

  // --- store bf16 result (C/D layout: M = r + 8*lhalf, N = lcol) ---
  int mrow = rbase + wave * 16 + lhalf * 8;
#pragma unroll
  for (int nt = 0; nt < 8; ++nt)
#pragma unroll
    for (int r = 0; r < 8; ++r)
      out[(size_t)(mrow + r) * HEAD + nt * 16 + lcol] = f2bf(acc[nt][r]);
}

// ---------------------------------------------------------------------------
// Kernel 2: causal flash attention, bf16 WMMA, f32 online softmax.
// grid.x: 2048/128 query tiles, grid.z: batch. 256 threads = 8 waves.
// Each wave owns a 16-row query strip.
// ---------------------------------------------------------------------------
__global__ __launch_bounds__(256) void attn_kernel(
    const unsigned short* __restrict__ q,
    const unsigned short* __restrict__ k,
    const unsigned short* __restrict__ v,
    float* __restrict__ out)
{
  __shared__ unsigned short Ks[32 * 128];    // [key][h]  row-major
  __shared__ unsigned short Vt[128 * 32];    // [h][key]  transposed
  __shared__ unsigned short Ps[8][16 * 32];  // per-wave P tile [m][key]

  const int b     = blockIdx.z;
  const int qbase = blockIdx.x * 128;
  const int tid   = threadIdx.x;
  const int wave  = tid >> 5;
  const int lane  = tid & 31;
  const int lhalf = lane >> 4;
  const int lcol  = lane & 15;
  const size_t bo = (size_t)b * SEQ * HEAD;

  // --- Q A-fragments held in registers: 4 chunks of K=32 over head dim ---
  v8u qf[4];
  {
    int row = qbase + wave * 16 + lcol;
    const unsigned int* qp = (const unsigned int*)(q + bo + (size_t)row * HEAD);
    int khalf = lhalf * 8;
#pragma unroll
    for (int c = 0; c < 4; ++c)
#pragma unroll
      for (int e2 = 0; e2 < 8; ++e2) {
        int kk = 32 * c + (e2 & 3) * 2 + (e2 >> 2) * 16 + khalf;
        qf[c][e2] = qp[kk >> 1];
      }
  }

  v8f acc[8] = {};
  float mrow[8], lrow[8];
#pragma unroll
  for (int r = 0; r < 8; ++r) { mrow[r] = -__builtin_inff(); lrow[r] = 0.f; }

  const int nblocks = (qbase + 127) / 32 + 1;         // causal frontier
  const int qmaxw   = qbase + wave * 16 + 15;
  const float scale = 0.03125f;                        // C**-0.5, C = 1024

  for (int j = 0; j < nblocks; ++j) {
    const int kbase = j * 32;

    // --- K tile: async global->LDS copy (bypasses VGPRs, ASYNCcnt) ---
    {
      const char* src = (const char*)(k + bo + (size_t)kbase * HEAD);
      char* dst = (char*)Ks;
      async_cp_b128(src + tid * 16, dst + tid * 16);
      async_cp_b128(src + 4096 + tid * 16, dst + 4096 + tid * 16);
    }
    // --- V tile: staged through VGPRs, transposed to [h][key] ---
    {
      const unsigned short* src = v + bo + (size_t)kbase * HEAD;
#pragma unroll
      for (int i2 = 0; i2 < 2; ++i2) {
        int f   = tid * 16 + i2 * 8;
        int key = f >> 7;
        int h   = f & 127;
        uint4 d = *(const uint4*)(src + (size_t)key * HEAD + h);
        Vt[(h + 0) * 32 + key] = (unsigned short)(d.x & 0xffff);
        Vt[(h + 1) * 32 + key] = (unsigned short)(d.x >> 16);
        Vt[(h + 2) * 32 + key] = (unsigned short)(d.y & 0xffff);
        Vt[(h + 3) * 32 + key] = (unsigned short)(d.y >> 16);
        Vt[(h + 4) * 32 + key] = (unsigned short)(d.z & 0xffff);
        Vt[(h + 5) * 32 + key] = (unsigned short)(d.z >> 16);
        Vt[(h + 6) * 32 + key] = (unsigned short)(d.w & 0xffff);
        Vt[(h + 7) * 32 + key] = (unsigned short)(d.w >> 16);
      }
    }
    // --- prefetch next tiles into cache while this block computes ---
    if (j + 1 < nblocks) {
      const char* kn = (const char*)(k + bo + (size_t)(kbase + 32) * HEAD);
      const char* vn = (const char*)(v + bo + (size_t)(kbase + 32) * HEAD);
      __builtin_prefetch(kn + tid * 32, 0, 1);
      __builtin_prefetch(vn + tid * 32, 0, 1);
    }
    wait_async_lds();
    __syncthreads();

    if (kbase <= qmaxw) {   // wave-uniform causal liveness
      // --- S = Q K^T : two 16x16 score tiles over 32 keys ---
      v8f s0 = {}, s1 = {};
#pragma unroll
      for (int c = 0; c < 4; ++c) {
        v8u b0, b1;
        const unsigned int* kp = (const unsigned int*)Ks;
        int kh = lhalf * 8 + 32 * c;
#pragma unroll
        for (int e2 = 0; e2 < 8; ++e2) {
          b0[e2] = kp[((lcol)      * HEAD + kh + (e2 & 3) * 2 + (e2 >> 2) * 16) >> 1];
          b1[e2] = kp[((lcol + 16) * HEAD + kh + (e2 & 3) * 2 + (e2 >> 2) * 16) >> 1];
        }
        s0 = wmma_bf16(qf[c], b0, s0);
        s1 = wmma_bf16(qf[c], b1, s1);
      }

      // --- scale + causal mask + online softmax (row groups of 16 lanes) ---
      float rmax[8];
#pragma unroll
      for (int r = 0; r < 8; ++r) {
        int qrow = qbase + wave * 16 + r + 8 * lhalf;
        float a0 = s0[r] * scale;
        float a1 = s1[r] * scale;
        if (kbase + lcol      > qrow) a0 = -__builtin_inff();
        if (kbase + 16 + lcol > qrow) a1 = -__builtin_inff();
        s0[r] = a0; s1[r] = a1;
        float mx = fmaxf(a0, a1);
#pragma unroll
        for (int off = 8; off; off >>= 1) mx = fmaxf(mx, __shfl_xor(mx, off, 32));
        rmax[r] = mx;
      }
#pragma unroll
      for (int r = 0; r < 8; ++r) {
        float mnew = fmaxf(mrow[r], rmax[r]);
        float al = __expf(mrow[r] - mnew);
        float p0 = __expf(s0[r] - mnew);
        float p1 = __expf(s1[r] - mnew);
        float ps = p0 + p1;
#pragma unroll
        for (int off = 8; off; off >>= 1) ps += __shfl_xor(ps, off, 32);
        lrow[r] = lrow[r] * al + ps;
        mrow[r] = mnew;
#pragma unroll
        for (int nt = 0; nt < 8; ++nt) acc[nt][r] *= al;
        int m = r + 8 * lhalf;
        Ps[wave][m * 32 + lcol]      = f2bf(p0);
        Ps[wave][m * 32 + 16 + lcol] = f2bf(p1);
      }

      // intra-wave LDS RAW: D-layout stores -> A-layout loads
      asm volatile("s_wait_dscnt 0" ::: "memory");

      // --- P A-fragment from this wave's LDS tile ---
      v8u pf;
      {
        const unsigned int* pp = (const unsigned int*)&Ps[wave][0];
        int khalf = lhalf * 8;
#pragma unroll
        for (int e2 = 0; e2 < 8; ++e2) {
          int kk = (e2 & 3) * 2 + (e2 >> 2) * 16 + khalf;
          pf[e2] = pp[(lcol * 32 + kk) >> 1];
        }
      }
      // --- acc += P V ---
#pragma unroll
      for (int nt = 0; nt < 8; ++nt) {
        v8u bf;
        const unsigned int* vp = (const unsigned int*)Vt;
        int n  = lcol + nt * 16;
        int kh = lhalf * 8;
#pragma unroll
        for (int e2 = 0; e2 < 8; ++e2)
          bf[e2] = vp[(n * 32 + kh + (e2 & 3) * 2 + (e2 >> 2) * 16) >> 1];
        acc[nt] = wmma_bf16(pf, bf, acc[nt]);
      }
    }
    __syncthreads();
  }

  // --- epilogue: out = acc / l (f32) ---
  int qrow0 = qbase + wave * 16 + 8 * lhalf;
#pragma unroll
  for (int r = 0; r < 8; ++r) {
    float inv = 1.0f / lrow[r];
    int row = qrow0 + r;
#pragma unroll
    for (int nt = 0; nt < 8; ++nt)
      out[bo + (size_t)row * HEAD + nt * 16 + lcol] = acc[nt][r] * inv;
  }
}

// ---------------------------------------------------------------------------
extern "C" void kernel_launch(void* const* d_in, const int* in_sizes, int n_in,
                              void* d_out, int out_size, void* d_ws, size_t ws_size,
                              hipStream_t stream) {
  const float* x  = (const float*)d_in[0];
  const float* Wq = (const float*)d_in[1];
  const float* Wk = (const float*)d_in[2];
  const float* Wv = (const float*)d_in[3];

  // bf16 Q/K/V intermediates in workspace: 3 * 8*2048*128 * 2B = 12 MB
  unsigned short* qws = (unsigned short*)d_ws;
  unsigned short* kws = qws + (size_t)BATCH * SEQ * HEAD;
  unsigned short* vws = kws + (size_t)BATCH * SEQ * HEAD;

  dim3 g1((BATCH * SEQ) / 128, 3, 1);
  proj_kernel<<<g1, 256, 0, stream>>>(x, Wq, Wk, Wv, qws, kws, vws);

  dim3 g2(SEQ / 128, 1, BATCH);
  attn_kernel<<<g2, 256, 0, stream>>>(qws, kws, vws, (float*)d_out);
}
